// ConvMultiHeadAttention_44289702756792
// MI455X (gfx1250) — matrix operand order repaired and checked by reading, not compiled, not run on hardware
//
#include <hip/hip_runtime.h>

#define NHH 8
#define DQKK 256
#define DVV 64
#define BB 8
#define TII 16
#define TOO 16
#define HH 32
#define WW 32

typedef _Float16 v8h  __attribute__((ext_vector_type(8)));
typedef _Float16 v16h __attribute__((ext_vector_type(16)));
typedef float    v8f  __attribute__((ext_vector_type(8)));

union V16U { v16h v; v8h h[2]; };

// ---------------------------------------------------------------------------
// K1: q/k linear projections per (b,h), scores, mask, softmax -> weights[b][h][o][i]
// ---------------------------------------------------------------------------
__global__ void __launch_bounds__(256)
attn_weights_kernel(const float* __restrict__ q, const float* __restrict__ k,
                    const int* __restrict__ mask,
                    const float* __restrict__ Wq, const float* __restrict__ bq,
                    const float* __restrict__ Wk, const float* __restrict__ bk,
                    float* __restrict__ weights)
{
    __shared__ float pq[TOO][DQKK];
    __shared__ float pk[TII][DQKK];
    __shared__ float sc[TOO][TII];
    const int bh = blockIdx.x;
    const int b = bh >> 3, h = bh & 7;
    const int d = threadIdx.x;          // 0..255 = feature within head

    {   // pq[o][d] = q[b,o,:] . Wq[:, h*256+d] + bq
        float acc[TOO];
#pragma unroll
        for (int o = 0; o < TOO; ++o) acc[o] = bq[h*DQKK + d];
        for (int e = 0; e < DQKK; ++e) {
            float wv = Wq[e*(NHH*DQKK) + h*DQKK + d];
#pragma unroll
            for (int o = 0; o < TOO; ++o) acc[o] += q[(b*TOO + o)*DQKK + e] * wv;
        }
#pragma unroll
        for (int o = 0; o < TOO; ++o) pq[o][d] = acc[o];
    }
    {   // pk[i][d]
        float acc[TII];
#pragma unroll
        for (int i = 0; i < TII; ++i) acc[i] = bk[h*DQKK + d];
        for (int e = 0; e < DQKK; ++e) {
            float wv = Wk[e*(NHH*DQKK) + h*DQKK + d];
#pragma unroll
            for (int i = 0; i < TII; ++i) acc[i] += k[(b*TII + i)*DQKK + e] * wv;
        }
#pragma unroll
        for (int i = 0; i < TII; ++i) pk[i][d] = acc[i];
    }
    __syncthreads();
    {   // scores (o,i) per thread
        const int o = threadIdx.x >> 4, i = threadIdx.x & 15;
        float s = 0.f;
        for (int dd = 0; dd < DQKK; ++dd) s += pq[o][dd] * pk[i][dd];
        s *= 0.0625f;                                   // 1/sqrt(256)
        if (mask[(b*TOO + o)*TII + i] == 0) s = -1e10f;
        sc[o][i] = s;
    }
    __syncthreads();
    if (threadIdx.x < TOO) {            // softmax over i, one row per thread
        const int o = threadIdx.x;
        float m = sc[o][0];
#pragma unroll
        for (int i = 1; i < TII; ++i) m = fmaxf(m, sc[o][i]);
        float e_[TII], sum = 0.f;
#pragma unroll
        for (int i = 0; i < TII; ++i) { e_[i] = __expf(sc[o][i] - m); sum += e_[i]; }
        const float inv = 1.f / sum;
#pragma unroll
        for (int i = 0; i < TII; ++i)
            weights[((b*NHH + h)*TOO + o)*TII + i] = e_[i] * inv;
    }
}

// ---------------------------------------------------------------------------
// K2: fold 1x1 projection into conv weights.
// weff[c][h][kk][ci] (f16) = sum_cv Wp[c, h*64+cv] * Wv[h*64+cv, ci, ky, kx]
// ---------------------------------------------------------------------------
__global__ void __launch_bounds__(256)
weff_kernel(const float* __restrict__ Wv, const float* __restrict__ Wp,
            _Float16* __restrict__ weff)
{
    const int idx = blockIdx.x * 256 + threadIdx.x;     // [c][h][kk][ci]
    const int ci = idx & 63;
    const int kk = (idx >> 6) % 9;
    const int h  = (idx / (64*9)) & 7;
    const int c  = idx / (64*9*8);
    float acc = 0.f;
    for (int cv = 0; cv < DVV; ++cv)
        acc += Wp[c*(NHH*DVV) + h*DVV + cv] * Wv[((h*DVV + cv)*DVV + ci)*9 + kk];
    weff[idx] = (_Float16)acc;
}

// K2b: beff[c] = bp[c] + sum_j Wp[c,j]*bv[j]   (softmax weights sum to 1)
__global__ void beff_kernel(const float* __restrict__ Wp, const float* __restrict__ bv,
                            const float* __restrict__ bp, float* __restrict__ beff)
{
    const int c = threadIdx.x;
    if (c < DVV) {
        float acc = bp[c];
        for (int j = 0; j < NHH*DVV; ++j) acc += Wp[c*(NHH*DVV) + j] * bv[j];
        beff[c] = acc;
    }
}

// ---------------------------------------------------------------------------
// K3: fused (attention value-mix) + (3x3 conv with folded weights) via WMMA.
// Grid: x = spatial slab (4 output rows), y = b*16+o. Block = 256 = 8 waves.
// Implicit GEMM: M=64 (out ch), N=128 (spatial), K=4608 = (h, ky*kx, ci).
// LDS holds head-mixed vbar tile [4 heads][6 rows][34 cols][72(pad) ci] f16.
// ---------------------------------------------------------------------------
__global__ void __launch_bounds__(256)
conv_attn_kernel(const float* __restrict__ v,        // [B,TI,64,32,32]
                 const float* __restrict__ weights,  // [B,NH,TO,TI]
                 const float* __restrict__ beff,     // [64]
                 const _Float16* __restrict__ weff,  // [c][h][kk][ci]
                 float* __restrict__ out)            // [B,TO,64,32,32]
{
    constexpr int CIP = 72;                          // padded ci stride (halves)
    extern __shared__ char smem[];
    float*    wlds = (float*)smem;                   // [NH][TI]
    _Float16* vbar = (_Float16*)(smem + 512);        // [4][6][34][CIP]

    const int bo = blockIdx.y;
    const int b  = bo >> 4;
    const int o  = bo & 15;
    const int y0 = blockIdx.x << 2;                  // 4 output rows per block

    const int tid = threadIdx.x;
    if (tid < NHH*TII) {
        const int h = tid >> 4, i = tid & 15;
        wlds[tid] = weights[((b*NHH + h)*TOO + o)*TII + i];
    }
    __syncthreads();

    const int wave  = tid >> 5;
    const int lane  = tid & 31;
    const int halfL = lane >> 4;
    const int l16   = lane & 15;
    const int mt    = wave & 3;                      // c-tile (M)
    const int g     = wave >> 2;                     // n-group

    v8f acc[4] = {};                                 // 4 N-tiles of 16x16

    int bOff[4], fOff[4];
#pragma unroll
    for (int j = 0; j < 4; ++j) {
        const int n  = ((g << 2) + j)*16 + l16;      // 0..127 within slab
        const int ny = n >> 5, nx = n & 31;
        bOff[j] = (ny*34 + nx)*CIP + (halfL << 4);   // + uniform site/ci per chunk
        fOff[j] = (y0 + ny)*WW + nx;
    }
    const int cA    = (mt << 4) + l16;               // A row = out channel
    const int aBase = cA*4608 + (halfL << 4);        // lane's K start (halves)

    for (int hp = 0; hp < 2; ++hp) {
        // ---- stage vbar for heads 4*hp..4*hp+3: mix 16 input frames ----
        for (int q8 = 0; q8 < 48; ++q8) {
            const int p   = wave*48 + q8;            // (row, ci) pair
            const int row = p >> 6;                  // 0..5 (y0-1 .. y0+4)
            const int ci  = p & 63;
            const int gy  = y0 - 1 + row;
            float vals[TII];
            if (gy >= 0 && gy < HH) {
                const float* vp = v + ((b*TII*DVV + ci)*1024 + gy*WW + lane);
#pragma unroll
                for (int i = 0; i < TII; ++i) vals[i] = vp[i*DVV*1024];
            } else {
#pragma unroll
                for (int i = 0; i < TII; ++i) vals[i] = 0.f;
            }
#pragma unroll
            for (int hl = 0; hl < 4; ++hl) {
                float a = 0.f;
#pragma unroll
                for (int i = 0; i < TII; ++i) a += wlds[((hp<<2) + hl)*TII + i] * vals[i];
                const int siteBase = (hl*6 + row)*34;
                vbar[(siteBase + lane + 1)*CIP + ci] = (_Float16)a;  // col = gx+1
                if (lane == 0)  vbar[(siteBase + 0 )*CIP + ci] = (_Float16)0.f;
                if (lane == 31) vbar[(siteBase + 33)*CIP + ci] = (_Float16)0.f;
            }
        }
        __syncthreads();

        // ---- 72 K-chunks of 32: (hl, kk, ci half) ----
        for (int t = 0; t < 72; ++t) {
            const int hl  = t / 18;
            const int r   = t % 18;
            const int kk  = r >> 1;
            const int ci0 = (r & 1) << 5;
            const int ky  = kk / 3, kx = kk - ky*3;
            const int h   = (hp << 2) + hl;

            // A fragment: Weff row cA, 2x contiguous 8-half chunks (contiguous k order, same as B)
            const _Float16* ap = weff + (aBase + h*576 + (kk << 6) + ci0);
            V16U au;
            au.h[0] = *(const v8h*)ap;               // K = 16*halfL + 0..7
            au.h[1] = *(const v8h*)(ap + 8);         // K = 16*halfL + 8..15

            const int u = ((hl*6 + ky)*34 + kx)*CIP + ci0;   // uniform per chunk
#pragma unroll
            for (int j = 0; j < 4; ++j) {
                const _Float16* bp2 = vbar + (u + bOff[j]);
                V16U bu;
                bu.h[0] = *(const v8h*)bp2;          // K = 16*halfL + 0..7
                bu.h[1] = *(const v8h*)(bp2 + 8);    // K = 16*halfL + 8..15
                acc[j] = __builtin_amdgcn_wmma_f32_16x16x32_f16(
                    false, au.v, false, bu.v, (short)0, acc[j], false, false);
            }
        }
        __syncthreads();
    }

    // ---- epilogue: add folded bias, store f32 output ----
    float* op = out + (long)bo * (DVV*1024);
#pragma unroll
    for (int r8 = 0; r8 < 8; ++r8) {
        const int c  = (mt << 4) + (halfL << 3) + r8;   // C/D: VGPR r -> M = r + 8*half
        const float be = beff[c];
#pragma unroll
        for (int j = 0; j < 4; ++j)
            op[c*1024 + fOff[j]] = acc[j][r8] + be;
    }
}

// ---------------------------------------------------------------------------
extern "C" void kernel_launch(void* const* d_in, const int* in_sizes, int n_in,
                              void* d_out, int out_size, void* d_ws, size_t ws_size,
                              hipStream_t stream)
{
    (void)in_sizes; (void)n_in; (void)out_size; (void)ws_size;
    const float* v    = (const float*)d_in[0];
    const float* k    = (const float*)d_in[1];
    const float* q    = (const float*)d_in[2];
    const int*   mask = (const int*)  d_in[3];
    const float* Wq   = (const float*)d_in[4];
    const float* bq   = (const float*)d_in[5];
    const float* Wk   = (const float*)d_in[6];
    const float* bk   = (const float*)d_in[7];
    const float* Wv   = (const float*)d_in[8];
    const float* bv   = (const float*)d_in[9];
    const float* Wp   = (const float*)d_in[10];
    const float* bp   = (const float*)d_in[11];
    float* out = (float*)d_out;

    // workspace layout: weights f32 (16384) | beff f32 (64) | weff f16 (294912)
    float*    wts  = (float*)d_ws;
    float*    beff = wts + 16384;
    _Float16* weff = (_Float16*)((char*)d_ws + 65792);

    attn_weights_kernel<<<64, 256, 0, stream>>>(q, k, mask, Wq, bq, Wk, bk, wts);
    weff_kernel<<<1152, 256, 0, stream>>>(Wv, Wp, weff);
    beff_kernel<<<1, 64, 0, stream>>>(Wp, bv, bp, beff);

    const size_t lds = 512 + (size_t)(4*6*34*72) * sizeof(_Float16);  // 118016 B
    conv_attn_kernel<<<dim3(8, 128), 256, lds, stream>>>(v, wts, beff, weff, out);
}
